// HeteroGraphSAGE_42915313221828
// MI455X (gfx1250) — matrix operand order repaired and checked by reading, not compiled, not run on hardware
//
#include <hip/hip_runtime.h>
#include <stdint.h>

typedef __attribute__((ext_vector_type(16))) __bf16 v16bf;
typedef __attribute__((ext_vector_type(8)))  float  v8f;

// ---------- helpers ----------

__device__ inline unsigned short f2bf(float f) {
  unsigned u = __float_as_uint(f);
  u += 0x7FFFu + ((u >> 16) & 1u);          // round-to-nearest-even
  return (unsigned short)(u >> 16);
}

__device__ inline v16bf frag16(const unsigned short* lo, const unsigned short* hi) {
  union { uint4 u[2]; v16bf v; } x;
  x.u[0] = *(const uint4*)lo;
  x.u[1] = *(const uint4*)hi;
  return x.v;
}

// hardware f32 atomic add at device scope (avoid CAS-loop fallback)
__device__ inline void atom_add_f32(float* p, float v) {
  asm volatile("global_atomic_add_f32 %0, %1, off scope:SCOPE_DEV"
               :: "v"((uint64_t)(uintptr_t)p), "v"(v) : "memory");
}

// ---------- kernels ----------

__global__ void k_zero(float* __restrict__ p, long n) {
  long i = (long)blockIdx.x * blockDim.x + threadIdx.x;
  if (i < n) p[i] = 0.0f;
}

// x[row] = emb[nid[row]]  (float4 per thread)
__global__ void k_gather(const float* __restrict__ emb, const int* __restrict__ nid,
                         float* __restrict__ x, int N) {
  long t = (long)blockIdx.x * blockDim.x + threadIdx.x;
  if (t >= (long)N * 16) return;
  int row = (int)(t >> 4);
  int c4  = (int)(t & 15) << 2;
  int s   = nid[row];
  *(float4*)(x + (long)row * 64 + c4) = *(const float4*)(emb + (long)s * 64 + c4);
}

__global__ void k_count(const int* __restrict__ dst, float* __restrict__ cnt, int E) {
  long t = (long)blockIdx.x * blockDim.x + threadIdx.x;
  if (t < E) atom_add_f32(&cnt[dst[t]], 1.0f);
}

// agg[dst[e]] += x[src[e]]   (float4 gathered, 4 scalar f32 atomics)
__global__ void k_scatter(const int* __restrict__ src, const int* __restrict__ dst,
                          const float* __restrict__ x, float* __restrict__ agg, int E) {
  long t = (long)blockIdx.x * blockDim.x + threadIdx.x;
  if (t >= (long)E * 16) return;
  int e  = (int)(t >> 4);
  int c4 = (int)(t & 15) << 2;
  int s = src[e], d = dst[e];
  float4 v = *(const float4*)(x + (long)s * 64 + c4);
  float* a = agg + (long)d * 64 + c4;
  atom_add_f32(a + 0, v.x);
  atom_add_f32(a + 1, v.y);
  atom_add_f32(a + 2, v.z);
  atom_add_f32(a + 3, v.w);
}

// Wt[n][k] = bf16( k<64 ? Wl[k][n] : Wr[k-64][n] )  -> [64][128] bf16, B-frag friendly
__global__ void k_wprep(const float* __restrict__ Wl, const float* __restrict__ Wr,
                        unsigned short* __restrict__ Wt) {
  int t = blockIdx.x * blockDim.x + threadIdx.x;
  if (t >= 64 * 128) return;
  int n = t >> 7, k = t & 127;
  float f = (k < 64) ? Wl[k * 64 + n] : Wr[(k - 64) * 64 + n];
  Wt[n * 128 + k] = f2bf(f);
}

// Fused: out = L2norm_rows( [agg/max(cnt,1) | xdst] @ Wt^T + bias ), optional ReLU.
// 256 threads = 8 waves; each wave owns a 16-row tile; 128 rows per block.
__global__ __launch_bounds__(256) void k_sage(
    const float* __restrict__ agg, const float* __restrict__ cnt,
    const float* __restrict__ xdst, const unsigned short* __restrict__ Wt,
    const float* __restrict__ bias, float* __restrict__ out, int N, int relu)
{
  __shared__ unsigned short sW[64 * 128];      // 16 KB weights (bf16, [n][k])
  __shared__ unsigned short sA[8][16 * 128];   // 32 KB activations (bf16, per-wave)

  int tid = threadIdx.x, wv = tid >> 5, lane = tid & 31;

  for (int t = tid; t < 64 * 128; t += 256) sW[t] = Wt[t];

  int rowbase = blockIdx.x * 128 + wv * 16;
  for (int t = lane; t < 16 * 128; t += 32) {
    int r = t >> 7, k = t & 127;
    int row = rowbase + r;
    float f = 0.0f;
    if (row < N) {
      if (k < 64) f = agg[(long)row * 64 + k] / fmaxf(cnt[row], 1.0f);
      else        f = xdst[(long)row * 64 + (k - 64)];
    }
    sA[wv][t] = f2bf(f);
  }
  __syncthreads();

  int half = lane >> 4, l15 = lane & 15;
  const unsigned short* aB = &sA[wv][l15 * 128 + (half << 3)];   // A: K base 0/8
  const unsigned short* bB = &sW[l15 * 128 + (half << 3)];       // B: K base 0/8

  v8f z = {0, 0, 0, 0, 0, 0, 0, 0};
  v8f acc[4] = {z, z, z, z};

#pragma unroll
  for (int ks = 0; ks < 4; ++ks) {
    v16bf a = frag16(aB + ks * 32, aB + ks * 32 + 16);
#pragma unroll
    for (int n = 0; n < 4; ++n) {
      const unsigned short* bp = bB + n * 2048 + ks * 32;
      v16bf b = frag16(bp, bp + 16);
      acc[n] = __builtin_amdgcn_wmma_f32_16x16x32_bf16(
          false, a, false, b, (short)0, acc[n], false, false);
    }
  }

  float bv[4];
#pragma unroll
  for (int n = 0; n < 4; ++n) bv[n] = bias[n * 16 + l15];

  int rowoff = half << 3;
#pragma unroll
  for (int v = 0; v < 8; ++v) {
    float val[4], ss = 0.0f;
#pragma unroll
    for (int n = 0; n < 4; ++n) { val[n] = acc[n][v] + bv[n]; ss += val[n] * val[n]; }
    ss += __shfl_xor(ss, 1);
    ss += __shfl_xor(ss, 2);
    ss += __shfl_xor(ss, 4);
    ss += __shfl_xor(ss, 8);       // sum over the 16-lane half that holds this row
    float nrm = fmaxf(sqrtf(ss), 1e-12f);
    int row = rowbase + rowoff + v;
    if (row < N) {
#pragma unroll
      for (int n = 0; n < 4; ++n) {
        float o = val[n] / nrm;
        if (relu) o = fmaxf(o, 0.0f);
        out[(long)row * 64 + n * 16 + l15] = o;
      }
    }
  }
}

// ---------- launch ----------

extern "C" void kernel_launch(void* const* d_in, const int* in_sizes, int n_in,
                              void* d_out, int out_size, void* d_ws, size_t ws_size,
                              hipStream_t stream) {
  (void)n_in; (void)out_size; (void)ws_size;

  const float* emb_user = (const float*)d_in[0];
  const float* emb_item = (const float*)d_in[1];
  const float* W1l_ui = (const float*)d_in[2];
  const float* b1_ui  = (const float*)d_in[3];
  const float* W1r_ui = (const float*)d_in[4];
  const float* W1l_iu = (const float*)d_in[5];
  const float* b1_iu  = (const float*)d_in[6];
  const float* W1r_iu = (const float*)d_in[7];
  const float* W2l_ui = (const float*)d_in[8];
  const float* b2_ui  = (const float*)d_in[9];
  const float* W2r_ui = (const float*)d_in[10];
  const float* W2l_iu = (const float*)d_in[11];
  const float* b2_iu  = (const float*)d_in[12];
  const float* W2r_iu = (const float*)d_in[13];
  const int* n_id_user = (const int*)d_in[14];
  const int* n_id_item = (const int*)d_in[15];
  const int* edge_ui   = (const int*)d_in[16];
  const int* edge_iu   = (const int*)d_in[17];

  int NU = in_sizes[0] / 64;
  int NI = in_sizes[1] / 64;
  int E  = in_sizes[16] / 2;
  int maxN = NU > NI ? NU : NI;

  float* U   = (float*)d_ws;                 // x_user, later h_user
  float* V   = U + (size_t)NU * 64;          // x_item
  float* AGG = V + (size_t)NI * 64;          // shared scatter accumulator
  float* cnt_ui = AGG + (size_t)maxN * 64;   // in-degree, item dst
  float* cnt_iu = cnt_ui + NI;               // in-degree, user dst
  unsigned short* Wt1ui = (unsigned short*)(cnt_iu + NU);
  unsigned short* Wt1iu = Wt1ui + 64 * 128;
  unsigned short* Wt2ui = Wt1iu + 64 * 128;
  unsigned short* Wt2iu = Wt2ui + 64 * 128;

  float* o_user = (float*)d_out;
  float* o_item = o_user + (size_t)NU * 64;  // doubles as h_item storage

  const int B = 256;
  auto cdiv = [](long a, long b) { return (int)((a + b - 1) / b); };

  k_wprep<<<32, B, 0, stream>>>(W1l_ui, W1r_ui, Wt1ui);
  k_wprep<<<32, B, 0, stream>>>(W1l_iu, W1r_iu, Wt1iu);
  k_wprep<<<32, B, 0, stream>>>(W2l_ui, W2r_ui, Wt2ui);
  k_wprep<<<32, B, 0, stream>>>(W2l_iu, W2r_iu, Wt2iu);

  k_gather<<<cdiv((long)NU * 16, B), B, 0, stream>>>(emb_user, n_id_user, U, NU);
  k_gather<<<cdiv((long)NI * 16, B), B, 0, stream>>>(emb_item, n_id_item, V, NI);

  k_zero<<<cdiv(NI, B), B, 0, stream>>>(cnt_ui, NI);
  k_zero<<<cdiv(NU, B), B, 0, stream>>>(cnt_iu, NU);
  k_count<<<cdiv(E, B), B, 0, stream>>>(edge_ui + E, cnt_ui, E);
  k_count<<<cdiv(E, B), B, 0, stream>>>(edge_iu + E, cnt_iu, E);

  // layer 1: h_item = relu(norm(mean_ui(x_user)@W1l_ui + b + x_item@W1r_ui))
  k_zero<<<cdiv((long)NI * 64, B), B, 0, stream>>>(AGG, (long)NI * 64);
  k_scatter<<<cdiv((long)E * 16, B), B, 0, stream>>>(edge_ui, edge_ui + E, U, AGG, E);
  k_sage<<<cdiv(NI, 128), B, 0, stream>>>(AGG, cnt_ui, V, Wt1ui, b1_ui, o_item, NI, 1);

  // layer 1: h_user (written in place over x_user)
  k_zero<<<cdiv((long)NU * 64, B), B, 0, stream>>>(AGG, (long)NU * 64);
  k_scatter<<<cdiv((long)E * 16, B), B, 0, stream>>>(edge_iu, edge_iu + E, V, AGG, E);
  k_sage<<<cdiv(NU, 128), B, 0, stream>>>(AGG, cnt_iu, U, Wt1iu, b1_iu, U, NU, 1);

  // layer 2: o_user = norm(mean_iu(h_item)@W2l_iu + b + h_user@W2r_iu)
  k_zero<<<cdiv((long)NU * 64, B), B, 0, stream>>>(AGG, (long)NU * 64);
  k_scatter<<<cdiv((long)E * 16, B), B, 0, stream>>>(edge_iu, edge_iu + E, o_item, AGG, E);
  k_sage<<<cdiv(NU, 128), B, 0, stream>>>(AGG, cnt_iu, U, Wt2iu, b2_iu, o_user, NU, 0);

  // layer 2: o_item (in place over h_item)
  k_zero<<<cdiv((long)NI * 64, B), B, 0, stream>>>(AGG, (long)NI * 64);
  k_scatter<<<cdiv((long)E * 16, B), B, 0, stream>>>(edge_ui, edge_ui + E, U, AGG, E);
  k_sage<<<cdiv(NI, 128), B, 0, stream>>>(AGG, cnt_ui, o_item, Wt2ui, b2_ui, o_item, NI, 0);
}
